// TopKSAE_50311246905715
// MI455X (gfx1250) — matrix operand order repaired and checked by reading, not compiled, not run on hardware
//
#include <hip/hip_runtime.h>

// ---------------------------------------------------------------------------
// TopK SAE forward for MI455X (gfx1250, wave32, WMMA + async global->LDS)
//   pre   = (x - b_dec) @ W_enc + b_enc        (f16 WMMA, f32 accum)
//   acts  = relu(topk_per_row(pre, 64))        (radix-select per row)
//   recon = acts @ W_dec + b_dec               (sparse gather-FMA)
// Fast path: pre-convert x-b_dec and W_enc^T to f16 in d_ws, then a
// double-buffered WMMA GEMM staged with GLOBAL_LOAD_ASYNC_TO_LDS_B128
// (no VGPR staging -> no spills; copy of tile i+1 overlaps WMMA on tile i).
// ---------------------------------------------------------------------------

typedef __attribute__((ext_vector_type(16))) _Float16 v16h;
typedef __attribute__((ext_vector_type(8)))  _Float16 v8h;
typedef __attribute__((ext_vector_type(4)))  _Float16 v4h;
typedef __attribute__((ext_vector_type(8)))  float    v8f;
typedef __attribute__((ext_vector_type(4)))  int      i32x4;

#define N_TOK   8192
#define DMODEL  2048
#define DSAE    16384
#define TOPK    64

#define BM 128
#define BN 128
#define BK 32
#define LDA 40   // BK + 8 pad; 80 bytes/row -> 16B aligned fragment loads
#define LDB 40

#define GLOBAL_AS __attribute__((address_space(1)))
#define LDS_AS    __attribute__((address_space(3)))

// ---- CDNA5 async global->LDS copy (ASYNCcnt path) --------------------------
__device__ __forceinline__ void async_copy_b128(const void* g, void* l) {
#if __has_builtin(__builtin_amdgcn_global_load_async_to_lds_b128)
  __builtin_amdgcn_global_load_async_to_lds_b128(
      (GLOBAL_AS i32x4*)g, (LDS_AS i32x4*)l, 0, 0);
#else
  unsigned loff = (unsigned)(size_t)(LDS_AS void*)l;  // byte offset in LDS
  asm volatile("global_load_async_to_lds_b128 %0, %1, off"
               :: "v"(loff), "v"(g) : "memory");
#endif
}
__device__ __forceinline__ void wait_async0() {
#if __has_builtin(__builtin_amdgcn_s_wait_asynccnt)
  __builtin_amdgcn_s_wait_asynccnt(0);
#else
  asm volatile("s_wait_asynccnt 0x0" ::: "memory");
#endif
}

// ===========================================================================
// Prep kernel A: x16[i][k] = (_Float16)(x[i][k] - b_dec[k])
// ===========================================================================
__global__ __launch_bounds__(256) void prep_x_f16(
    const float* __restrict__ x, const float* __restrict__ b_dec,
    _Float16* __restrict__ x16)
{
  const size_t total4 = (size_t)N_TOK * DMODEL / 4;
  for (size_t i = (size_t)blockIdx.x * 256 + threadIdx.x; i < total4;
       i += (size_t)gridDim.x * 256) {
    float4 f = ((const float4*)x)[i];
    int col = (int)((i * 4) & (DMODEL - 1));
    float4 bd = *(const float4*)(&b_dec[col]);
    v4h o;
    o[0] = (_Float16)(f.x - bd.x);
    o[1] = (_Float16)(f.y - bd.y);
    o[2] = (_Float16)(f.z - bd.z);
    o[3] = (_Float16)(f.w - bd.w);
    ((v4h*)x16)[i] = o;
  }
}

// ===========================================================================
// Prep kernel B: Wt[n][k] = (_Float16)W_enc[k][n]   (64x64 tiles via LDS)
// grid = (DSAE/64, DMODEL/64), block = 256
// ===========================================================================
__global__ __launch_bounds__(256) void prep_wenc_f16t(
    const float* __restrict__ W_enc, _Float16* __restrict__ Wt)
{
  __shared__ _Float16 tile[64 * 66];   // stride 66 f16 = 33 dwords (odd)
  const int t  = threadIdx.x;
  const int n0 = blockIdx.x * 64;      // sae dim
  const int k0 = blockIdx.y * 64;      // model dim
#pragma unroll
  for (int i = 0; i < 16; ++i) {
    int idx = i * 256 + t;             // 0..4095
    int r = idx >> 6;                  // k-local
    int c = idx & 63;                  // n-local
    tile[c * 66 + r] = (_Float16)W_enc[(size_t)(k0 + r) * DSAE + n0 + c];
  }
  __syncthreads();
#pragma unroll
  for (int i = 0; i < 8; ++i) {        // paired b32 stores
    int idx = i * 256 + t;             // 0..2047
    int rr = idx >> 5;                 // n-local
    int cp = (idx & 31) * 2;           // k-local pair
    unsigned lo = (unsigned)__builtin_bit_cast(unsigned short, tile[rr * 66 + cp]);
    unsigned hi = (unsigned)__builtin_bit_cast(unsigned short, tile[rr * 66 + cp + 1]);
    *(unsigned*)(&Wt[(size_t)(n0 + rr) * DMODEL + k0 + cp]) = lo | (hi << 16);
  }
}

// ===========================================================================
// Kernel 1 (fast): encode GEMM, f16 inputs, async-to-LDS double buffering
// grid = (DSAE/BN, N_TOK/BM), block = 256 (8 waves, 2x4 wave grid)
// ===========================================================================
__global__ __launch_bounds__(256) void sae_encode_wmma_f16(
    const _Float16* __restrict__ x16,   // [N_TOK, DMODEL]  (x - b_dec)
    const _Float16* __restrict__ Wt,    // [DSAE, DMODEL]   (W_enc^T)
    const float* __restrict__ b_enc,    // [DSAE]
    float* __restrict__ pre_out)        // [N_TOK, DSAE]
{
  __shared__ __attribute__((aligned(16))) _Float16 As[2][BM * LDA]; // [m][k]
  __shared__ __attribute__((aligned(16))) _Float16 Bs[2][BN * LDB]; // [n][k]

  const int tid  = threadIdx.x;
  const int lane = tid & 31;
  const int wave = tid >> 5;
  const int wm   = wave & 1;
  const int wn   = wave >> 1;
  const int bm0  = blockIdx.y * BM;
  const int bn0  = blockIdx.x * BN;

  v8f acc[4][2];
#pragma unroll
  for (int i = 0; i < 4; ++i)
#pragma unroll
    for (int j = 0; j < 2; ++j) acc[i][j] = (v8f){};

  // staging geometry: 2 b128 chunks per thread for each of A and B
  // (128 rows x 4 chunks of 8 f16 per tile)
  int rs[2], cs[2];
#pragma unroll
  for (int s = 0; s < 2; ++s) {
    int idx = s * 256 + tid;           // 0..511
    rs[s] = idx >> 2;                  // row 0..127
    cs[s] = (idx & 3) * 8;             // k element offset 0,8,16,24
  }

  const int r_in  = lane & 15;
  const int khalf = (lane >> 4) << 3;  // 0 or 8
  const int k0f   = (lane >> 4) << 4;  // 0 or 16

  auto stage_async = [&](int kk, int buf) {
#pragma unroll
    for (int s = 0; s < 2; ++s) {
      async_copy_b128(&x16[(size_t)(bm0 + rs[s]) * DMODEL + kk + cs[s]],
                      &As[buf][rs[s] * LDA + cs[s]]);
      async_copy_b128(&Wt [(size_t)(bn0 + rs[s]) * DMODEL + kk + cs[s]],
                      &Bs[buf][rs[s] * LDB + cs[s]]);
    }
  };

  stage_async(0, 0);
  wait_async0();
  __syncthreads();

  const int NIT = DMODEL / BK;         // 64
  for (int it = 0; it < NIT; ++it) {
    const int cur = it & 1;
    // kick off copy of next tile into the other buffer (overlaps WMMA below)
    if (it + 1 < NIT) stage_async((it + 1) * BK, cur ^ 1);

    // ---- fragments per documented CDNA5 16-bit layouts --------------------
    v16h afrag[4];
#pragma unroll
    for (int i = 0; i < 4; ++i) {
      int row = wm * 64 + i * 16 + r_in;
      v8h lo = *(const v8h*)(&As[cur][row * LDA + khalf]);
      v8h hi = *(const v8h*)(&As[cur][row * LDA + khalf + 16]);
      afrag[i] = __builtin_shufflevector(lo, hi, 0,1,2,3,4,5,6,7,
                                                 8,9,10,11,12,13,14,15);
    }
    v16h bfrag[2];
#pragma unroll
    for (int j = 0; j < 2; ++j) {
      int n = wn * 32 + j * 16 + (lane & 15);
      v8h lo = *(const v8h*)(&Bs[cur][n * LDB + khalf]);
      v8h hi = *(const v8h*)(&Bs[cur][n * LDB + khalf + 16]);
      bfrag[j] = __builtin_shufflevector(lo, hi, 0,1,2,3,4,5,6,7,
                                                 8,9,10,11,12,13,14,15);
    }
#pragma unroll
    for (int i = 0; i < 4; ++i)
#pragma unroll
      for (int j = 0; j < 2; ++j)
        acc[i][j] = __builtin_amdgcn_wmma_f32_16x16x32_f16(
            false, afrag[i], false, bfrag[j],
            (short)0, acc[i][j], false, false);

    if (it + 1 < NIT) {
      wait_async0();      // next buffer fully in LDS (this wave's copies)
      __syncthreads();    // all waves' copies done + all done reading cur
    }
  }

  // ---- epilogue: add b_enc, store pre ------------------------------------
  const int n_in = lane & 15;
  const int mb   = (lane >> 4) << 3;
#pragma unroll
  for (int j = 0; j < 2; ++j) {
    int col  = bn0 + wn * 32 + j * 16 + n_in;
    float be = b_enc[col];
#pragma unroll
    for (int i = 0; i < 4; ++i) {
      int row0 = bm0 + wm * 64 + i * 16 + mb;
#pragma unroll
      for (int v = 0; v < 8; ++v)
        pre_out[(size_t)(row0 + v) * DSAE + col] = acc[i][j][v] + be;
    }
  }
}

// ===========================================================================
// Kernel 1 (fallback, ws too small): f32 inputs, convert while staging
// ===========================================================================
__global__ __launch_bounds__(256) void sae_encode_wmma_f32(
    const float* __restrict__ x, const float* __restrict__ W_enc,
    const float* __restrict__ b_enc, const float* __restrict__ b_dec,
    float* __restrict__ pre_out)
{
  __shared__ __attribute__((aligned(16))) _Float16 As[BM * LDA];
  __shared__ __attribute__((aligned(16))) _Float16 Bs[BN * LDB];

  const int tid  = threadIdx.x;
  const int lane = tid & 31;
  const int wave = tid >> 5;
  const int wm   = wave & 1;
  const int wn   = wave >> 1;
  const int bm0  = blockIdx.y * BM;
  const int bn0  = blockIdx.x * BN;

  v8f acc[4][2];
#pragma unroll
  for (int i = 0; i < 4; ++i)
#pragma unroll
    for (int j = 0; j < 2; ++j) acc[i][j] = (v8f){};

  const int r_in  = lane & 15;
  const int khalf = (lane >> 4) << 3;
  const int k0f   = (lane >> 4) << 4;

  for (int kk = 0; kk < DMODEL; kk += BK) {
#pragma unroll
    for (int it = 0; it < 4; ++it) {
      int idx = it * 256 + tid;
      int r   = idx >> 3;
      int c4  = (idx & 7) << 2;
      const float4 xv = *(const float4*)(&x[(size_t)(bm0 + r) * DMODEL + kk + c4]);
      const float4 bd = *(const float4*)(&b_dec[kk + c4]);
      v4h o;
      o[0] = (_Float16)(xv.x - bd.x);
      o[1] = (_Float16)(xv.y - bd.y);
      o[2] = (_Float16)(xv.z - bd.z);
      o[3] = (_Float16)(xv.w - bd.w);
      *(v4h*)(&As[r * LDA + c4]) = o;
    }
#pragma unroll
    for (int it = 0; it < 4; ++it) {
      int idx = it * 256 + tid;
      int kr  = idx >> 5;
      int c4  = (idx & 31) << 2;
      const float4 wv = *(const float4*)(&W_enc[(size_t)(kk + kr) * DSAE + bn0 + c4]);
      Bs[(c4 + 0) * LDB + kr] = (_Float16)wv.x;
      Bs[(c4 + 1) * LDB + kr] = (_Float16)wv.y;
      Bs[(c4 + 2) * LDB + kr] = (_Float16)wv.z;
      Bs[(c4 + 3) * LDB + kr] = (_Float16)wv.w;
    }
    __syncthreads();

    v16h afrag[4];
#pragma unroll
    for (int i = 0; i < 4; ++i) {
      int row = wm * 64 + i * 16 + r_in;
      v8h lo = *(const v8h*)(&As[row * LDA + khalf]);
      v8h hi = *(const v8h*)(&As[row * LDA + khalf + 16]);
      afrag[i] = __builtin_shufflevector(lo, hi, 0,1,2,3,4,5,6,7,
                                                 8,9,10,11,12,13,14,15);
    }
    v16h bfrag[2];
#pragma unroll
    for (int j = 0; j < 2; ++j) {
      int n = wn * 32 + j * 16 + (lane & 15);
      v8h lo = *(const v8h*)(&Bs[n * LDB + khalf]);
      v8h hi = *(const v8h*)(&Bs[n * LDB + khalf + 16]);
      bfrag[j] = __builtin_shufflevector(lo, hi, 0,1,2,3,4,5,6,7,
                                                 8,9,10,11,12,13,14,15);
    }
#pragma unroll
    for (int i = 0; i < 4; ++i)
#pragma unroll
      for (int j = 0; j < 2; ++j)
        acc[i][j] = __builtin_amdgcn_wmma_f32_16x16x32_f16(
            false, afrag[i], false, bfrag[j],
            (short)0, acc[i][j], false, false);
    __syncthreads();
  }

  const int n_in = lane & 15;
  const int mb   = (lane >> 4) << 3;
#pragma unroll
  for (int j = 0; j < 2; ++j) {
    int col  = bn0 + wn * 32 + j * 16 + n_in;
    float be = b_enc[col];
#pragma unroll
    for (int i = 0; i < 4; ++i) {
      int row0 = bm0 + wm * 64 + i * 16 + mb;
#pragma unroll
      for (int v = 0; v < 8; ++v)
        pre_out[(size_t)(row0 + v) * DSAE + col] = acc[i][j][v] + be;
    }
  }
}

// ===========================================================================
// Kernel 2: per-row exact top-64 (radix select) + ReLU + sparse decode
// ===========================================================================
__device__ __forceinline__ unsigned f2key(unsigned b) {
  return b ^ (((int)b >> 31) | 0x80000000u);
}
__device__ __forceinline__ float key2f(unsigned u) {
  unsigned b = (u & 0x80000000u) ? (u ^ 0x80000000u) : ~u;
  return __uint_as_float(b);
}

__global__ __launch_bounds__(256) void sae_topk_decode(
    const float* __restrict__ W_dec,  // [DSAE, DMODEL]
    const float* __restrict__ b_dec,  // [DMODEL]
    float* __restrict__ acts,         // in: pre, out: sparse relu acts
    float* __restrict__ recon)        // [N_TOK, DMODEL]
{
  extern __shared__ unsigned smem[];
  unsigned* uvals  = smem;                    // [DSAE]
  unsigned* hist   = uvals + DSAE;            // [256]
  unsigned* mcnt   = hist + 256;              // [257]
  unsigned* sctrl  = mcnt + 257;              // [2]
  int*      selcol = (int*)(sctrl + 2);       // [TOPK]
  float*    selval = (float*)(selcol + TOPK); // [TOPK]

  const int row = blockIdx.x;
  const int t   = threadIdx.x;
  float* prow = &acts[(size_t)row * DSAE];

#pragma unroll 4
  for (int i4 = 0; i4 < 16; ++i4) {
    int e = t * 64 + i4 * 4;
    const float4 f = *(const float4*)(&prow[e]);
    uvals[e + 0] = f2key(__float_as_uint(f.x));
    uvals[e + 1] = f2key(__float_as_uint(f.y));
    uvals[e + 2] = f2key(__float_as_uint(f.z));
    uvals[e + 3] = f2key(__float_as_uint(f.w));
  }
  if (t == 0) { sctrl[0] = 0u; sctrl[1] = TOPK; }
  __syncthreads();

  for (int shift = 24; shift >= 0; shift -= 8) {
    hist[t] = 0;
    __syncthreads();
    const unsigned pf     = sctrl[0];
    const unsigned himask = (shift == 24) ? 0u : (0xFFFFFFFFu << (shift + 8));
    for (int i = 0; i < 64; ++i) {
      unsigned u = uvals[t * 64 + i];
      if ((u & himask) == pf) atomicAdd(&hist[(u >> shift) & 255u], 1u);
    }
    __syncthreads();
    if (t == 0) {
      unsigned kk = sctrl[1], cum = 0;
      int b = 255;
      for (; b > 0; --b) { cum += hist[b]; if (cum >= kk) break; }
      if (cum < kk) cum += hist[0];
      sctrl[1] = kk - (cum - hist[b]);
      sctrl[0] = pf | ((unsigned)b << shift);
    }
    __syncthreads();
  }
  const unsigned thr = sctrl[0];

  int m = 0;
  for (int i = 0; i < 64; ++i) m += (uvals[t * 64 + i] >= thr) ? 1 : 0;
  mcnt[t] = (unsigned)m;
  __syncthreads();
  if (t == 0) {
    unsigned run = 0;
    for (int j = 0; j < 256; ++j) { unsigned c = mcnt[j]; mcnt[j] = run; run += c; }
    mcnt[256] = run;
  }
  __syncthreads();

  unsigned pos = mcnt[t];
  for (int i4 = 0; i4 < 16; ++i4) {
    int e = t * 64 + i4 * 4;
    float4 out;
    float* o = &out.x;
#pragma unroll
    for (int j = 0; j < 4; ++j) {
      unsigned u = uvals[e + j];
      bool keep  = (u >= thr);
      float rv   = fmaxf(key2f(u), 0.0f);
      o[j]       = keep ? rv : 0.0f;
      if (keep) {
        if (pos < TOPK) { selcol[pos] = e + j; selval[pos] = rv; }
        ++pos;
      }
    }
    *(float4*)(&prow[e]) = out;
  }
  __syncthreads();
  const int nsel = (mcnt[256] < TOPK) ? (int)mcnt[256] : TOPK;

  // ---- sparse decode: each thread owns 8 contiguous recon columns --------
  const int c0 = t * 8;
  float4 r0 = *(const float4*)(&b_dec[c0]);
  float4 r1 = *(const float4*)(&b_dec[c0 + 4]);
  for (int j = 0; j < nsel; ++j) {
    const float v = selval[j];
    const float* wd = &W_dec[(size_t)selcol[j] * DMODEL];
    const float4 w0 = *(const float4*)(&wd[c0]);
    const float4 w1 = *(const float4*)(&wd[c0 + 4]);
    r0.x = fmaf(v, w0.x, r0.x); r0.y = fmaf(v, w0.y, r0.y);
    r0.z = fmaf(v, w0.z, r0.z); r0.w = fmaf(v, w0.w, r0.w);
    r1.x = fmaf(v, w1.x, r1.x); r1.y = fmaf(v, w1.y, r1.y);
    r1.z = fmaf(v, w1.z, r1.z); r1.w = fmaf(v, w1.w, r1.w);
  }
  *(float4*)(&recon[(size_t)row * DMODEL + c0])     = r0;
  *(float4*)(&recon[(size_t)row * DMODEL + c0 + 4]) = r1;
}

// ===========================================================================
extern "C" void kernel_launch(void* const* d_in, const int* in_sizes, int n_in,
                              void* d_out, int out_size, void* d_ws, size_t ws_size,
                              hipStream_t stream) {
  const float* x     = (const float*)d_in[0];  // [8192, 2048]
  const float* W_enc = (const float*)d_in[1];  // [2048, 16384]
  const float* b_enc = (const float*)d_in[2];  // [16384]
  const float* W_dec = (const float*)d_in[3];  // [16384, 2048]
  const float* b_dec = (const float*)d_in[4];  // [2048]

  float* recon = (float*)d_out;                               // [8192, 2048]
  float* acts  = (float*)d_out + (size_t)N_TOK * DMODEL;      // [8192, 16384]

  dim3 g1(DSAE / BN, N_TOK / BM);
  const size_t x16_bytes = (size_t)N_TOK * DMODEL * sizeof(_Float16);
  const size_t wt_bytes  = (size_t)DSAE * DMODEL * sizeof(_Float16);

  if (ws_size >= x16_bytes + wt_bytes) {
    _Float16* x16 = (_Float16*)d_ws;
    _Float16* Wt  = (_Float16*)((char*)d_ws + x16_bytes);
    prep_x_f16<<<4096, 256, 0, stream>>>(x, b_dec, x16);
    prep_wenc_f16t<<<dim3(DSAE / 64, DMODEL / 64), 256, 0, stream>>>(W_enc, Wt);
    sae_encode_wmma_f16<<<g1, 256, 0, stream>>>(x16, Wt, b_enc, acts);
  } else {
    sae_encode_wmma_f32<<<g1, 256, 0, stream>>>(x, W_enc, b_enc, b_dec, acts);
  }

  size_t smem2 = (size_t)(DSAE + 256 + 257 + 2 + TOPK + TOPK) * sizeof(unsigned);
  sae_topk_decode<<<N_TOK, 256, smem2, stream>>>(W_dec, b_dec, acts, recon);
}
